// SolverVAE_50723563766141
// MI455X (gfx1250) — matrix operand order repaired and checked by reading, not compiled, not run on hardware
//
#include <hip/hip_runtime.h>
#include <hip/hip_bf16.h>

typedef __attribute__((ext_vector_type(16))) _Float16 v16h;
typedef __attribute__((ext_vector_type(8)))  _Float16 v8h;
typedef __attribute__((ext_vector_type(8)))  float    v8f;
typedef __attribute__((ext_vector_type(2)))  float    v2f;

#define BATCH 2048
#define XDIM  256
#define YDIM  128
#define MDIM  64
#define H1DIM 1024
#define H2DIM 512

// Use CDNA5 async global->LDS DMA for GEMM A-slab staging.
#define USE_ASYNC_STAGING 1

__device__ __forceinline__ void wait_asynccnt0() {
#if __has_builtin(__builtin_amdgcn_s_wait_asynccnt)
    __builtin_amdgcn_s_wait_asynccnt(0);
#else
    asm volatile("s_wait_asynccnt 0x0" ::: "memory");
#endif
}

// ---------------------------------------------------------------- utilities

__device__ __forceinline__ unsigned pcg_hash(unsigned v) {
    v = v * 747796405u + 2891336453u;
    unsigned w = ((v >> ((v >> 28u) + 4u)) ^ v) * 277803737u;
    return (w >> 22u) ^ w;
}
__device__ __forceinline__ float u01(unsigned h) {
    float u = (float)(h >> 8) * (1.0f / 16777216.0f);
    return fminf(fmaxf(u, 1e-6f), 1.0f - 1e-6f);
}
__device__ __forceinline__ float rng_u(unsigned idx, unsigned salt, unsigned ctr) {
    return u01(pcg_hash(idx ^ pcg_hash(salt + ctr * 0x9E3779B9u)));
}

// Marsaglia–Tsang gamma sampler (bounded iterations, deterministic)
__device__ float gamma_sample(float shape, unsigned idx, unsigned salt) {
    float a = fmaxf(shape, 1e-6f);
    float boost = 1.0f;
    if (a < 1.0f) {
        float u = rng_u(idx, salt, 101u);
        boost = powf(u, 1.0f / a);
        a += 1.0f;
    }
    float d = a - (1.0f / 3.0f);
    float c = 1.0f / (3.0f * sqrtf(d));
    float res = d;  // fallback ~ mean
    for (int it = 0; it < 8; ++it) {
        float u1 = rng_u(idx, salt, 2u * it);
        float u2 = rng_u(idx, salt, 2u * it + 1u);
        float n  = sqrtf(-2.0f * logf(u1)) * cosf(6.28318530718f * u2);
        float w  = 1.0f + c * n;
        if (w <= 0.0f) continue;
        float v  = w * w * w;
        float u3 = rng_u(idx, salt, 1000u + it);
        if (logf(u3) < 0.5f * n * n + d - d * v + d * logf(v)) { res = d * v; break; }
    }
    return boost * res;
}

__device__ __forceinline__ float apply_act(float v, int act) {
    if (act == 1) return v * (1.0f / (1.0f + expf(-v)));             // silu
    if (act == 2) return fmaxf(v, 0.0f) + log1pf(expf(-fabsf(v)));   // softplus
    return v;
}

// ------------------------------------------------ weight transpose + f16 cvt
// src: f32 [K,N] row-major  ->  dst: f16 [N,K] row-major (both dirs coalesced)
__global__ __launch_bounds__(256)
void transpose_cvt_f16(const float* __restrict__ src, _Float16* __restrict__ dst,
                       int K, int N) {
    __shared__ _Float16 tile[32][33];
    const int kb = blockIdx.y * 32, nb = blockIdx.x * 32;
    const int tx = threadIdx.x, ty = threadIdx.y;       // 32 x 8
    for (int i = ty; i < 32; i += 8)
        tile[i][tx] = (_Float16)src[(size_t)(kb + i) * N + nb + tx];
    __syncthreads();
    for (int i = ty; i < 32; i += 8)
        dst[(size_t)(nb + i) * K + kb + tx] = tile[tx][i];
}

// ---------------------------------------------------------------- f32 -> f16
__global__ void cvt_f16(const float* __restrict__ in, _Float16* __restrict__ out, int n) {
    int i = blockIdx.x * blockDim.x + threadIdx.x;
    if (i < n) out[i] = (_Float16)in[i];
}

// ------------------------------------------------------------ concat [y|x] f16
__global__ void concat_yx_f16(const float* __restrict__ y, const float* __restrict__ x,
                              _Float16* __restrict__ out, int n) {
    int i = blockIdx.x * blockDim.x + threadIdx.x;
    if (i >= n) return;
    int b = i / (YDIM + XDIM), j = i % (YDIM + XDIM);
    out[i] = (_Float16)((j < YDIM) ? y[b * YDIM + j] : x[b * XDIM + (j - YDIM)]);
}

// ---------------------------------------------------------------- encoder GEMM
// Out = act( A(f16,[B,K]) @ W(f16,[N,K] transposed) + bias ), f32 accumulate.
// Block = 256 threads (8 waves); A-slab (16 rows x K) staged in LDS once via
// async global->LDS DMA; each wave computes TWO 16x16 tiles (N % 32 == 0
// always holds for this model), reusing its A fragment across both WMMAs.
__global__ __launch_bounds__(256)
void gemm_wmma_f16(const _Float16* __restrict__ Ah, const _Float16* __restrict__ Wt,
                   const float* __restrict__ bias, float* __restrict__ OutF,
                   _Float16* __restrict__ OutH, int K, int N, int act) {
    __shared__ _Float16 Asm[16 * 1024];                 // 32 KB, K <= 1024

    const int tid  = threadIdx.x;
    const int wave = tid >> 5, lane = tid & 31;
    const int half = lane >> 4, r = lane & 15;
    const int m0   = blockIdx.y * 16;

    // cooperative staging of the A slab (shared by all 8 waves)
    {
        const int kc = K >> 3;                // 8-half (16B) chunks per row
        const int chunks = 16 * kc;
        for (int c = tid; c < chunks; c += 256) {
            int row = c / kc, off = (c - row * kc) << 3;
            const _Float16* gsrc = Ah + (size_t)(m0 + row) * K + off;
            _Float16* ldst = Asm + row * K + off;
#if USE_ASYNC_STAGING
            // per-lane 16B DMA: LDS[vdst] <- MEM[vaddr]  (ASYNCcnt-tracked)
            unsigned ldsoff = (unsigned)(size_t)ldst;   // low 32 bits = LDS offset
            asm volatile("global_load_async_to_lds_b128 %0, %1, off"
                         :: "v"(ldsoff), "v"(gsrc) : "memory");
#else
            *(v8h*)ldst = *(const v8h*)gsrc;
#endif
        }
#if USE_ASYNC_STAGING
        wait_asynccnt0();
#endif
    }
    __syncthreads();

    const int ct0 = blockIdx.x * 16 + wave * 2;         // first column tile
    if ((ct0 * 16) < N) {
        const int n0 = ct0 * 16 + r;
        const int n1 = n0 + 16;
        const _Float16* Arow = Asm + r * K;
        const _Float16* B0 = Wt + (size_t)n0 * K;
        const _Float16* B1 = Wt + (size_t)n1 * K;

        v8f acc0 = {}, acc1 = {};
        for (int k0 = 0; k0 < K; k0 += 32) {
            v16h a, b0, b1;
            const v8h alo = *(const v8h*)(Arow + k0 + (half << 3));        // ds_load_b128
            const v8h ahi = *(const v8h*)(Arow + k0 + 16 + (half << 3));
            const v8h bl0 = *(const v8h*)(B0 + k0 + (half << 3));          // global_load_b128
            const v8h bh0 = *(const v8h*)(B0 + k0 + 16 + (half << 3));
            const v8h bl1 = *(const v8h*)(B1 + k0 + (half << 3));
            const v8h bh1 = *(const v8h*)(B1 + k0 + 16 + (half << 3));
#pragma unroll
            for (int e = 0; e < 8; ++e) {
                a[e]  = alo[e]; a[8 + e]  = ahi[e];
                b0[e] = bl0[e]; b0[8 + e] = bh0[e];
                b1[e] = bl1[e]; b1[8 + e] = bh1[e];
            }
            acc0 = __builtin_amdgcn_wmma_f32_16x16x32_f16(false, a, false, b0,
                                                          (short)0, acc0, false, false);
            acc1 = __builtin_amdgcn_wmma_f32_16x16x32_f16(false, a, false, b1,
                                                          (short)0, acc1, false, false);
        }
        const float bc0 = bias[n0];
        const float bc1 = bias[n1];
#pragma unroll
        for (int v = 0; v < 8; ++v) {                   // C/D 16x16 f32 layout
            int mo = m0 + (half << 3) + v;
            float v0 = apply_act(acc0[v] + bc0, act);
            float v1 = apply_act(acc1[v] + bc1, act);
            if (OutF) { OutF[(size_t)mo * N + n0] = v0;
                        OutF[(size_t)mo * N + n1] = v1; }
            if (OutH) { OutH[(size_t)mo * N + n0] = (_Float16)v0;
                        OutH[(size_t)mo * N + n1] = (_Float16)v1; }
        }
    }
}

// ---------------------------------------------------------------- LP solver
// One block (256 threads, 8 waves) per batch element. Sampling fused in.
__global__ __launch_bounds__(256)
void lp_solve_kernel(const float* __restrict__ Wlog, const float* __restrict__ hs,
                     const float* __restrict__ hr,   const float* __restrict__ qs,
                     const float* __restrict__ qr,   float* __restrict__ yhat) {
    __shared__ float A_s[MDIM * YDIM];     // 32 KB: sampled constraint matrix
    __shared__ float Hm[YDIM * YDIM];      // 64 KB: Newton Hessian / Cholesky
    __shared__ float xv[YDIM], gv[YDIM], dxv[YDIM], cv[YDIM];
    __shared__ float bv[MDIM], sv[MDIM], isv[MDIM], is2[MDIM];
    __shared__ float red[256];
    __shared__ float scal[4];

    const int tid  = threadIdx.x;
    const int bi   = blockIdx.x;
    const float BIG = 3.4e38f;

    // --- sample A = sigmoid(logit + logit(u))  (relaxed Bernoulli, tau=1)
    for (int idx = tid; idx < MDIM * YDIM; idx += 256) {
        unsigned gidx = (unsigned)bi * (MDIM * YDIM) + idx;
        float logit = Wlog[(size_t)bi * (MDIM * YDIM) + idx];
        float u = rng_u(gidx, 0x9E3779B9u, 0u);
        A_s[idx] = 1.0f / (1.0f + expf(-(logit + logf(u) - log1pf(-u))));
    }
    // --- sample b = gamma(hs)/hr ; c = clamp(gamma(qs)/qr, 1e-3)
    if (tid < MDIM) {
        unsigned gidx = (unsigned)bi * MDIM + tid;
        bv[tid] = gamma_sample(hs[(size_t)bi * MDIM + tid], gidx, 0x85EBCA6Bu)
                  / hr[(size_t)bi * MDIM + tid];
    }
    if (tid < YDIM) {
        unsigned gidx = (unsigned)bi * YDIM + tid;
        cv[tid] = fmaxf(gamma_sample(qs[(size_t)bi * YDIM + tid], gidx, 0xC2B2AE35u)
                        / qr[(size_t)bi * YDIM + tid], 1e-3f);
    }
    __syncthreads();

    // --- strictly feasible init: x0 = max(0.5*min(b/(A@1 + 1e-6)), 1e-6)
    if (tid < MDIM) {
        float rs = 0.0f;
        const float* Ar = &A_s[tid * YDIM];
        for (int j = 0; j < YDIM; ++j) rs += Ar[j];
        red[tid] = bv[tid] / (rs + 1e-6f);
    }
    __syncthreads();
    if (tid == 0) {
        float mn = red[0];
        for (int m = 1; m < MDIM; ++m) mn = fminf(mn, red[m]);
        scal[0] = fmaxf(0.5f * mn, 1e-6f);
    }
    __syncthreads();
    if (tid < YDIM) xv[tid] = scal[0];
    __syncthreads();

    float t = 1.0f;
    for (int outer = 0; outer < 4; ++outer) {
        for (int inner = 0; inner < 2; ++inner) {
            // s = b - A x ; 1/s ; 1/s^2
            if (tid < MDIM) {
                float acc = 0.0f;
                const float* Ar = &A_s[tid * YDIM];
                for (int j = 0; j < YDIM; ++j) acc += Ar[j] * xv[j];
                float s = bv[tid] - acc;
                sv[tid] = s; float inv = 1.0f / s;
                isv[tid] = inv; is2[tid] = inv * inv;
            }
            __syncthreads();
            // g = t*c - 1/x + A^T (1/s)
            if (tid < YDIM) {
                float acc = 0.0f;
                for (int m = 0; m < MDIM; ++m) acc += A_s[m * YDIM + tid] * isv[m];
                gv[tid] = t * cv[tid] - 1.0f / xv[tid] + acc;
            }
            __syncthreads();

            // Hm = (A^T diag(1/s^2)) A  via f32 WMMA 16x16x4; 64 tiles, 8 waves
#if __has_builtin(__builtin_amdgcn_wmma_f32_16x16x4_f32)
            {
                const int wave = tid >> 5, lane = tid & 31;
                const int half = lane >> 4, r = lane & 15;
                for (int tile = wave; tile < 64; tile += 8) {
                    int ti = tile >> 3, tj = tile & 7;
                    v8f acc = {};
                    for (int k0 = 0; k0 < MDIM; k0 += 4) {
                        v2f av, bvf;
#pragma unroll
                        for (int v = 0; v < 2; ++v) {
                            int k = k0 + (half << 1) + v;
                            av[v]  = A_s[k * YDIM + ti * 16 + r] * is2[k];
                            bvf[v] = A_s[k * YDIM + tj * 16 + r];
                        }
                        acc = __builtin_amdgcn_wmma_f32_16x16x4_f32(
                                  false, av, false, bvf, (short)0, acc, false, false);
                    }
#pragma unroll
                    for (int v = 0; v < 8; ++v) {
                        int mi = ti * 16 + (half << 3) + v;
                        Hm[mi * YDIM + tj * 16 + r] = acc[v];
                    }
                }
            }
#else
            for (int idx = tid; idx < YDIM * YDIM; idx += 256) {
                int i = idx >> 7, j = idx & 127;
                float acc = 0.0f;
                for (int m = 0; m < MDIM; ++m)
                    acc += A_s[m * YDIM + i] * is2[m] * A_s[m * YDIM + j];
                Hm[idx] = acc;
            }
#endif
            __syncthreads();
            if (tid < YDIM) {
                float ix = 1.0f / xv[tid];
                Hm[tid * YDIM + tid] += ix * ix + 1e-8f;
            }
            __syncthreads();

            // in-place Cholesky (full-square trailing update keeps symmetry)
            for (int k = 0; k < YDIM; ++k) {
                if (tid == 0) Hm[k * YDIM + k] = sqrtf(fmaxf(Hm[k * YDIM + k], 1e-30f));
                __syncthreads();
                float dk = Hm[k * YDIM + k];
                for (int i = k + 1 + tid; i < YDIM; i += 256) Hm[i * YDIM + k] /= dk;
                __syncthreads();
                int nt = YDIM - 1 - k;
                for (int idx = tid; idx < nt * nt; idx += 256) {
                    int i = k + 1 + idx / nt, j = k + 1 + idx % nt;
                    Hm[i * YDIM + j] -= Hm[i * YDIM + k] * Hm[j * YDIM + k];
                }
                __syncthreads();
            }

            // triangular solves by wave 0, solution striped across 32 lanes
            if (tid < 32) {
                float yreg[4] = {0, 0, 0, 0}, dreg[4] = {0, 0, 0, 0};
                for (int i = 0; i < YDIM; ++i) {       // L y = -g
                    float part = 0.0f;
#pragma unroll
                    for (int jj = 0; jj < 4; ++jj) {
                        int j = jj * 32 + tid;
                        if (j < i) part += Hm[i * YDIM + j] * yreg[jj];
                    }
                    for (int off = 16; off; off >>= 1) part += __shfl_xor(part, off, 32);
                    float yi = (-gv[i] - part) / Hm[i * YDIM + i];
                    if ((i & 31) == tid) yreg[i >> 5] = yi;
                }
                for (int i = YDIM - 1; i >= 0; --i) {  // L^T dx = y
                    float part = 0.0f;
#pragma unroll
                    for (int jj = 0; jj < 4; ++jj) {
                        int j = jj * 32 + tid;
                        if (j > i) part += Hm[j * YDIM + i] * dreg[jj];
                    }
                    for (int off = 16; off; off >>= 1) part += __shfl_xor(part, off, 32);
                    float yi = __shfl(yreg[i >> 5], i & 31, 32);
                    float di = (yi - part) / Hm[i * YDIM + i];
                    if ((i & 31) == tid) dreg[i >> 5] = di;
                }
#pragma unroll
                for (int jj = 0; jj < 4; ++jj) dxv[jj * 32 + tid] = dreg[jj];
            }
            __syncthreads();

            // step_x = min over dx<0 of -x/dx
            if (tid < YDIM) {
                float dx = dxv[tid];
                red[tid] = (dx < 0.0f) ? (-xv[tid] / dx) : BIG;
            } else red[tid] = BIG;
            __syncthreads();
            for (int sft = 128; sft > 0; sft >>= 1) {
                if (tid < sft) red[tid] = fminf(red[tid], red[tid + sft]);
                __syncthreads();
            }
            if (tid == 0) scal[1] = red[0];
            __syncthreads();
            // step_s = min over (A dx)>0 of s/(A dx)
            if (tid < MDIM) {
                float acc = 0.0f;
                const float* Ar = &A_s[tid * YDIM];
                for (int j = 0; j < YDIM; ++j) acc += Ar[j] * dxv[j];
                red[tid] = (acc > 0.0f) ? (sv[tid] / acc) : BIG;
            } else red[tid] = BIG;
            __syncthreads();
            for (int sft = 128; sft > 0; sft >>= 1) {
                if (tid < sft) red[tid] = fminf(red[tid], red[tid + sft]);
                __syncthreads();
            }
            if (tid == 0) scal[2] = red[0];
            __syncthreads();
            float step = fminf(1.0f, 0.95f * fminf(scal[1], scal[2]));
            if (tid < YDIM) xv[tid] += step * dxv[tid];
            __syncthreads();
        }
        t *= 30.0f;
    }
    if (tid < YDIM) yhat[(size_t)bi * YDIM + tid] = xv[tid];
}

// ---------------------------------------------------------------- launch

static inline void launch_gemm(const _Float16* Ah, const _Float16* Wt, const float* bias,
                               float* OutF, _Float16* OutH, int K, int N, int act,
                               hipStream_t s) {
    dim3 grid((N + 255) / 256, BATCH / 16);
    gemm_wmma_f16<<<grid, dim3(256), 0, s>>>(Ah, Wt, bias, OutF, OutH, K, N, act);
}
static inline void launch_tr(const float* src, _Float16* dst, int K, int N, hipStream_t s) {
    transpose_cvt_f16<<<dim3(N / 32, K / 32), dim3(32, 8), 0, s>>>(src, dst, K, N);
}

extern "C" void kernel_launch(void* const* d_in, const int* in_sizes, int n_in,
                              void* d_out, int out_size, void* d_ws, size_t ws_size,
                              hipStream_t stream) {
    const float* y = (const float*)d_in[0];
    const float* x = (const float*)d_in[1];
    // dict keys sorted: W, fc1, fc2, hr, hs, qr, qs ; each [w, b]
    const float** P = (const float**)&d_in[2];    // prior params
    const float** Q = (const float**)&d_in[16];   // recog params
    // 0 W_w, 1 W_b, 2 fc1_w, 3 fc1_b, 4 fc2_w, 5 fc2_b,
    // 6 hr_w, 7 hr_b, 8 hs_w, 9 hs_b, 10 qr_w, 11 qr_b, 12 qs_w, 13 qs_b

    float* out = (float*)d_out;
    const size_t SZ_W = (size_t)BATCH * MDIM * YDIM;
    const size_t SZ_M = (size_t)BATCH * MDIM;
    const size_t SZ_Y = (size_t)BATCH * YDIM;
    float* o_Wp  = out;
    float* o_hsp = o_Wp  + SZ_W;
    float* o_hrp = o_hsp + SZ_M;
    float* o_qsp = o_hrp + SZ_M;
    float* o_qrp = o_qsp + SZ_Y;
    float* o_Wq  = o_qrp + SZ_Y;
    float* o_hsq = o_Wq  + SZ_W;
    float* o_hrq = o_hsq + SZ_M;
    float* o_qsq = o_hrq + SZ_M;
    float* o_qrq = o_qsq + SZ_Y;
    float* o_yh  = o_qrq + SZ_Y;

    // workspace carve (f16 elements); weight buffers reused across encoders
    _Float16* w16 = (_Float16*)d_ws;
    _Float16* xh    = w16;                                    // B*256
    _Float16* cat   = xh    + (size_t)BATCH * XDIM;           // B*384
    _Float16* h1h   = cat   + (size_t)BATCH * (YDIM + XDIM);  // B*1024
    _Float16* h2h   = h1h   + (size_t)BATCH * H1DIM;          // B*512
    _Float16* fc1t  = h2h   + (size_t)BATCH * H2DIM;          // 384*1024 max
    _Float16* fc2t  = fc1t  + (size_t)(YDIM + XDIM) * H1DIM;  // 1024*512
    _Float16* Wht   = fc2t  + (size_t)H1DIM * H2DIM;          // 512*8192
    _Float16* hrt   = Wht   + (size_t)H2DIM * MDIM * YDIM;    // 512*64
    _Float16* hst   = hrt   + (size_t)H2DIM * MDIM;
    _Float16* qrt   = hst   + (size_t)H2DIM * MDIM;           // 512*128
    _Float16* qst   = qrt   + (size_t)H2DIM * YDIM;

    // ---- input conversions
    {
        int n = BATCH * XDIM;
        cvt_f16<<<(n + 255) / 256, 256, 0, stream>>>(x, xh, n);
        n = BATCH * (YDIM + XDIM);
        concat_yx_f16<<<(n + 255) / 256, 256, 0, stream>>>(y, x, cat, n);
    }

    // ---- prior encoder (input x)
    launch_tr(P[2], fc1t, XDIM,  H1DIM, stream);
    launch_tr(P[4], fc2t, H1DIM, H2DIM, stream);
    launch_tr(P[0], Wht,  H2DIM, MDIM * YDIM, stream);
    launch_tr(P[6], hrt,  H2DIM, MDIM, stream);
    launch_tr(P[8], hst,  H2DIM, MDIM, stream);
    launch_tr(P[10], qrt, H2DIM, YDIM, stream);
    launch_tr(P[12], qst, H2DIM, YDIM, stream);

    launch_gemm(xh,  fc1t, P[3],  nullptr, h1h, XDIM,  H1DIM, 1, stream);
    launch_gemm(h1h, fc2t, P[5],  nullptr, h2h, H1DIM, H2DIM, 1, stream);
    launch_gemm(h2h, Wht,  P[1],  o_Wp,  nullptr, H2DIM, MDIM * YDIM, 0, stream);
    launch_gemm(h2h, hst,  P[9],  o_hsp, nullptr, H2DIM, MDIM, 2, stream);
    launch_gemm(h2h, hrt,  P[7],  o_hrp, nullptr, H2DIM, MDIM, 2, stream);
    launch_gemm(h2h, qst,  P[13], o_qsp, nullptr, H2DIM, YDIM, 2, stream);
    launch_gemm(h2h, qrt,  P[11], o_qrp, nullptr, H2DIM, YDIM, 2, stream);

    // ---- recog encoder (input [y|x]); weight buffers reused (stream-ordered)
    launch_tr(Q[2], fc1t, YDIM + XDIM, H1DIM, stream);
    launch_tr(Q[4], fc2t, H1DIM, H2DIM, stream);
    launch_tr(Q[0], Wht,  H2DIM, MDIM * YDIM, stream);
    launch_tr(Q[6], hrt,  H2DIM, MDIM, stream);
    launch_tr(Q[8], hst,  H2DIM, MDIM, stream);
    launch_tr(Q[10], qrt, H2DIM, YDIM, stream);
    launch_tr(Q[12], qst, H2DIM, YDIM, stream);

    launch_gemm(cat, fc1t, Q[3],  nullptr, h1h, YDIM + XDIM, H1DIM, 1, stream);
    launch_gemm(h1h, fc2t, Q[5],  nullptr, h2h, H1DIM, H2DIM, 1, stream);
    launch_gemm(h2h, Wht,  Q[1],  o_Wq,  nullptr, H2DIM, MDIM * YDIM, 0, stream);
    launch_gemm(h2h, hst,  Q[9],  o_hsq, nullptr, H2DIM, MDIM, 2, stream);
    launch_gemm(h2h, hrt,  Q[7],  o_hrq, nullptr, H2DIM, MDIM, 2, stream);
    launch_gemm(h2h, qst,  Q[13], o_qsq, nullptr, H2DIM, YDIM, 2, stream);
    launch_gemm(h2h, qrt,  Q[11], o_qrq, nullptr, H2DIM, YDIM, 2, stream);

    // ---- sample + interior-point solve (fused), one block per batch element
    lp_solve_kernel<<<dim3(BATCH), dim3(256), 0, stream>>>(
        o_Wp, o_hsp, o_hrp, o_qsp, o_qrp, o_yh);

    (void)in_sizes; (void)n_in; (void)out_size; (void)ws_size;
}
